// DDIRegularizer_57896159150306
// MI455X (gfx1250) — matrix operand order repaired and checked by reading, not compiled, not run on hardware
//
#include <hip/hip_runtime.h>
#include <hip/hip_bf16.h>

// ---------------------------------------------------------------------------
// DDI regularizer:  out = mean_b( x_b^T U x_b ) / max(sum(U), 1)
//   U = triu(sym-binarized ddi, k=1), x = drug_probs (4096 x 2048 f32)
// Strategy:
//   1) build Ut (U transposed, f16, exact 0/1) + popcount normalizer
//   2) convert x to f16 (xh)
//   3) fused WMMA GEMM, 2x4 register-blocked per wave:
//      8 x v_wmma_f32_16x16x32_f16 per K-step fed by 12 fragment loads
//      (1.5 loads per WMMA), then elementwise vs f32 x + wave reduce
//   4) deterministic single-block reduction + scale by 1/(norm * B)
// ---------------------------------------------------------------------------

#define B_DIM 4096
#define D_DIM 2048

// GEMM blocking: wave tile 32x64 (2x4 of 16x16), block = 8 waves as 2(M)x4(N)
// -> block tile 64x256; grid = (2048/256, 4096/64) = (8, 64); 4096 waves.
#define GRID_X (D_DIM / 256)          // 8
#define GRID_Y (B_DIM / 64)           // 64
#define NPARTIALS (GRID_X * GRID_Y * 8)  // 4096

typedef __attribute__((ext_vector_type(16))) _Float16 v16h;
typedef __attribute__((ext_vector_type(8)))  _Float16 v8h;
typedef __attribute__((ext_vector_type(4)))  _Float16 v4h;
typedef __attribute__((ext_vector_type(8)))  float    v8f;

// ---------------------------------------------------------------------------
__global__ void ddi_init(unsigned int* norm_count) {
    if (threadIdx.x == 0) *norm_count = 0u;
}

// ---------------------------------------------------------------------------
// Ut[n][k] = (k < n) && (ddi[k][n] > 0 || ddi[n][k] > 0)   (f16 exact 0/1)
// Counts set bits into norm_count (integer atomics -> deterministic).
__global__ __launch_bounds__(256)
void ddi_build_upper(const float* __restrict__ ddi,
                     _Float16* __restrict__ Ut,
                     unsigned int* __restrict__ norm_count) {
    size_t idx = (size_t)blockIdx.x * 256 + threadIdx.x;   // over D*D
    int n = (int)(idx / D_DIM);
    int k = (int)(idx % D_DIM);
    bool val = (k < n) &&
               ((ddi[(size_t)k * D_DIM + n] > 0.0f) ||
                (ddi[(size_t)n * D_DIM + k] > 0.0f));
    Ut[idx] = val ? (_Float16)1.0f : (_Float16)0.0f;
    unsigned long long mask = __ballot(val);               // wave32: low 32 bits
    if ((threadIdx.x & 31) == 0)
        atomicAdd(norm_count, (unsigned int)__popcll(mask));
}

// ---------------------------------------------------------------------------
// xh = (f16) x, 4 elements per thread.
__global__ __launch_bounds__(256)
void ddi_cvt_x(const float* __restrict__ x, _Float16* __restrict__ xh) {
    size_t idx = ((size_t)blockIdx.x * 256 + threadIdx.x) * 4;
    const float4 v = *(const float4*)(x + idx);
    v4h o;
    o[0] = (_Float16)v.x; o[1] = (_Float16)v.y;
    o[2] = (_Float16)v.z; o[3] = (_Float16)v.w;
    *(v4h*)(xh + idx) = o;
}

// ---------------------------------------------------------------------------
// Fragment loaders (contiguous 16B pairs per lane).
// A 16x32 f16: lanes 0-15 hold K={0..7,16..23}, lanes 16-31 K={8..15,24..31}.
__device__ __forceinline__ v16h load_frag_a(const _Float16* row, int k0, int hi) {
    v8h a0 = *(const v8h*)(row + k0 + hi * 8);
    v8h a1 = *(const v8h*)(row + k0 + 16 + hi * 8);
    return __builtin_shufflevector(a0, a1, 0,1,2,3,4,5,6,7,8,9,10,11,12,13,14,15);
}
// B 32x16 f16: lane = column, VGPRs sweep K; Ut stores K contiguous per column.
__device__ __forceinline__ v16h load_frag_b(const _Float16* col, int k0, int hi) {
    v8h b0 = *(const v8h*)(col + k0 + hi * 8);
    v8h b1 = *(const v8h*)(col + k0 + 16 + hi * 8);
    return __builtin_shufflevector(b0, b1, 0,1,2,3,4,5,6,7,8,9,10,11,12,13,14,15);
}

__device__ __forceinline__ v8f wmma_f16(v16h a, v16h b, v8f c) {
    return __builtin_amdgcn_wmma_f32_16x16x32_f16(false, a, false, b,
                                                  (short)0, c, false, false);
}

// ---------------------------------------------------------------------------
// Fused GEMM + elementwise + per-wave reduce. 2x4 register blocking.
__global__ __launch_bounds__(256)
void ddi_gemm(const _Float16* __restrict__ xh,
              const _Float16* __restrict__ Ut,
              const float*    __restrict__ x,
              float*          __restrict__ partials) {
    const int lane  = threadIdx.x & 31;
    const int wave  = threadIdx.x >> 5;    // 0..7
    const int lo    = lane & 15;
    const int hi    = lane >> 4;
    const int waveM = wave & 1;            // 0..1
    const int waveN = wave >> 1;           // 0..3

    const int m0 = blockIdx.y * 64 + waveM * 32;    // wave rows m0..m0+31
    const int n0 = blockIdx.x * 256 + waveN * 64;   // wave cols n0..n0+63

    const _Float16* aRow[2];
    const _Float16* bCol[4];
#pragma unroll
    for (int mi = 0; mi < 2; ++mi)
        aRow[mi] = xh + (size_t)(m0 + mi * 16 + lo) * D_DIM;
#pragma unroll
    for (int ni = 0; ni < 4; ++ni)
        bCol[ni] = Ut + (size_t)(n0 + ni * 16 + lo) * D_DIM;

    v8f acc[2][4];
#pragma unroll
    for (int mi = 0; mi < 2; ++mi)
#pragma unroll
        for (int ni = 0; ni < 4; ++ni)
            acc[mi][ni] = (v8f){};

    for (int k0 = 0; k0 < D_DIM; k0 += 32) {
        v16h a[2], b[4];
#pragma unroll
        for (int mi = 0; mi < 2; ++mi) a[mi] = load_frag_a(aRow[mi], k0, hi);
#pragma unroll
        for (int ni = 0; ni < 4; ++ni) b[ni] = load_frag_b(bCol[ni], k0, hi);
#pragma unroll
        for (int mi = 0; mi < 2; ++mi)
#pragma unroll
            for (int ni = 0; ni < 4; ++ni)
                acc[mi][ni] = wmma_f16(a[mi], b[ni], acc[mi][ni]);
    }

    // C layout per tile: VGPR i -> row (tileM + hi*8 + i), lane -> col (tileN + lo).
    float partial = 0.0f;
#pragma unroll
    for (int mi = 0; mi < 2; ++mi) {
#pragma unroll
        for (int i = 0; i < 8; ++i) {
            const size_t r = (size_t)(m0 + mi * 16 + hi * 8 + i) * D_DIM;
#pragma unroll
            for (int ni = 0; ni < 4; ++ni)
                partial += acc[mi][ni][i] * x[r + (n0 + ni * 16 + lo)];
        }
    }
#pragma unroll
    for (int off = 16; off > 0; off >>= 1)
        partial += __shfl_down(partial, off, 32);
    if (lane == 0)
        partials[((size_t)blockIdx.y * GRID_X + blockIdx.x) * 8 + wave] = partial;
}

// ---------------------------------------------------------------------------
// Deterministic final reduction: fixed-order grid-stride + LDS tree.
__global__ __launch_bounds__(256)
void ddi_reduce(const float* __restrict__ partials,
                const unsigned int* __restrict__ norm_count,
                float* __restrict__ out) {
    __shared__ float smem[256];
    float s = 0.0f;
    for (int i = threadIdx.x; i < NPARTIALS; i += 256)
        s += partials[i];
    smem[threadIdx.x] = s;
    __syncthreads();
    for (int stride = 128; stride > 0; stride >>= 1) {
        if ((int)threadIdx.x < stride)
            smem[threadIdx.x] += smem[threadIdx.x + stride];
        __syncthreads();
    }
    if (threadIdx.x == 0) {
        float norm = (float)(*norm_count);
        if (norm < 1.0f) norm = 1.0f;
        out[0] = smem[0] / norm / (float)B_DIM;
    }
}

// ---------------------------------------------------------------------------
extern "C" void kernel_launch(void* const* d_in, const int* in_sizes, int n_in,
                              void* d_out, int out_size, void* d_ws, size_t ws_size,
                              hipStream_t stream) {
    const float* x   = (const float*)d_in[0];   // (B, D) drug_probs
    const float* ddi = (const float*)d_in[1];   // (D, D) ddi_matrix
    float* out = (float*)d_out;

    // Workspace layout (~24.1 MB total):
    char* ws = (char*)d_ws;
    unsigned int* norm_count = (unsigned int*)ws;                       // 4 B
    float*    partials = (float*)(ws + 256);                            // 16 KB
    _Float16* Ut = (_Float16*)(ws + 256 + 16384 + 256);                 // 8 MB
    _Float16* xh = (_Float16*)(ws + 256 + 16384 + 256 +
                               (size_t)D_DIM * D_DIM * 2);              // 16 MB

    ddi_init<<<1, 64, 0, stream>>>(norm_count);

    ddi_build_upper<<<(D_DIM * D_DIM) / 256, 256, 0, stream>>>(ddi, Ut, norm_count);

    ddi_cvt_x<<<(B_DIM * D_DIM) / (256 * 4), 256, 0, stream>>>(x, xh);

    dim3 grid(GRID_X, GRID_Y);   // (8, 64), 8 waves per block
    ddi_gemm<<<grid, 256, 0, stream>>>(xh, Ut, x, partials);

    ddi_reduce<<<1, 256, 0, stream>>>(partials, norm_count, out);
}
